// ChannelReductionAttention_76690936037434
// MI455X (gfx1250) — matrix operand order repaired and checked by reading, not compiled, not run on hardware
//
#include <hip/hip_runtime.h>
#include <hip/hip_bf16.h>
#include <math.h>

#define B_      8
#define C_      256
#define N_      4096   // 64*64 tokens
#define M_      1024   // 32*32 pooled tokens
#define NHEADS  8
#define DHEAD   32
#define SCALE_  0.17677669529663687f   // 1/sqrt(32)
#define LN_EPS_ 1e-5f

typedef __attribute__((ext_vector_type(16))) __bf16 v16bf;
typedef __attribute__((ext_vector_type(8)))  float  v8f;

union Frag16 {
  uint4 u4[2];   // 32 bytes
  v16bf v;       // 16 bf16
};

__device__ __forceinline__ unsigned short bfbits(float x) {
  union { __bf16 b; unsigned short u; } c; c.b = (__bf16)x; return c.u;
}
__device__ __forceinline__ float bf2f(unsigned short u) {
  union { __bf16 b; unsigned short u; } c; c.u = u; return (float)c.b;
}
__device__ __forceinline__ float wave_sum(float v) {
  for (int o = 16; o > 0; o >>= 1) v += __shfl_xor(v, o, 32);
  return v;
}
__device__ __forceinline__ float wave_max(float v) {
  for (int o = 16; o > 0; o >>= 1) v = fmaxf(v, __shfl_xor(v, o, 32));
  return v;
}
__device__ __forceinline__ float wave_min(float v) {
  for (int o = 16; o > 0; o >>= 1) v = fminf(v, __shfl_xor(v, o, 32));
  return v;
}

// ---------------------------------------------------------------------------
// f32 -> bf16 weight conversion
// ---------------------------------------------------------------------------
__global__ void f32_to_bf16_kernel(const float* __restrict__ in,
                                   unsigned short* __restrict__ out, int n) {
  int i = blockIdx.x * blockDim.x + threadIdx.x;
  if (i < n) out[i] = bfbits(in[i]);
}

// ---------------------------------------------------------------------------
// 2x2 average pool; x [B,C,64,64] -> poolBf tokens [B*M, C] bf16 (row-major)
// ---------------------------------------------------------------------------
__global__ void pool_kernel(const float* __restrict__ x,
                            unsigned short* __restrict__ poolBf) {
  int tid = blockIdx.x * blockDim.x + threadIdx.x;      // (b, c, m), m fastest
  if (tid >= B_ * C_ * M_) return;
  int m = tid & (M_ - 1);
  int c = (tid >> 10) & (C_ - 1);
  int b = tid >> 18;
  int mi = m >> 5, mj = m & 31;
  const float* p = x + (((size_t)b * C_ + c) * 64 + mi * 2) * 64 + mj * 2;
  float avg = 0.25f * (p[0] + p[1] + p[64] + p[65]);
  poolBf[((size_t)b * M_ + m) * C_ + c] = bfbits(avg);
}

// ---------------------------------------------------------------------------
// Generic bf16 WMMA GEMM:  out[r, o] = sum_c A[r, c] * W[o, c]  (+ bias[o])
//   A: [rows, K] bf16 row-major.   W: [cols, K] bf16 row-major (weightsᵀ form)
// One wave per 16x64 D macro-tile: one A fragment feeds 4 B fragments and
// 4 WMMAs per 32-deep K step (4x A reuse, 4 WMMAs per 10 b128 loads).
//   TRANS store:  out[b][o][r_local] (ldOut stride per o)
//   normal store: out[r][o]          (ldOut stride per r)
// ---------------------------------------------------------------------------
template <typename OutT, bool TRANS, bool BIAS>
__global__ void gemm_bf16_wmma(const unsigned short* __restrict__ A,
                               const unsigned short* __restrict__ W,
                               const float* __restrict__ bias,
                               OutT* __restrict__ out,
                               int rows, int K, int cols,
                               int rows_per_batch, long long out_batch_stride,
                               int ldOut) {
  const int gw = (int)((blockIdx.x * blockDim.x + threadIdx.x) >> 5);
  const int colTiles = cols >> 6;            // 64-wide macro tiles
  const int rowTiles = rows >> 4;
  if (gw >= rowTiles * colTiles) return;
  const int rowT = gw / colTiles;
  const int colT = gw % colTiles;
  const int lane = threadIdx.x & 31;
  const int l16  = lane & 15;
  const bool hi  = lane >= 16;

  // A frag (16x32 bf16): lanes 0-15 hold K 0-7 & 16-23, lanes 16-31 hold 8-15 & 24-31
  const unsigned short* ap  = A + (size_t)(rowT * 16 + l16) * K + (hi ? 8 : 0);
  // B frag (32x16 bf16): lane = column o, lanes 0-15 hold K 0-7 & 16-23, 16-31 hold 8-15 & 24-31
  const unsigned short* wp0 = W + (size_t)(colT * 64 + l16) * K + (hi ? 8 : 0);

  v8f acc[4] = {};
  for (int kk = 0; kk < K; kk += 32) {
    Frag16 a;
    a.u4[0] = *(const uint4*)(ap + kk);
    a.u4[1] = *(const uint4*)(ap + kk + 16);
    if (kk + 32 < K) __builtin_prefetch(ap + kk + 32, 0, 1);
#pragma unroll
    for (int j = 0; j < 4; ++j) {
      const unsigned short* wp = wp0 + (size_t)(j * 16) * K + kk;
      Frag16 b;
      b.u4[0] = *(const uint4*)(wp);
      b.u4[1] = *(const uint4*)(wp + 16);
      acc[j] = __builtin_amdgcn_wmma_f32_16x16x32_bf16(false, a.v, false, b.v,
                                                       (short)0, acc[j], false, false);
    }
  }

#pragma unroll
  for (int j = 0; j < 4; ++j) {
    const int col = colT * 64 + j * 16 + l16;
    const float bv = BIAS ? bias[col] : 0.0f;
#pragma unroll
    for (int r = 0; r < 8; ++r) {
      const int row = rowT * 16 + r + (hi ? 8 : 0);  // D layout: lanes 16-31 = rows 8-15
      float vv = acc[j][r] + bv;
      if constexpr (TRANS) {
        const int bb = row / rows_per_batch;
        const int rl = row - bb * rows_per_batch;
        OutT* p = out + (size_t)bb * out_batch_stride + (size_t)col * ldOut + rl;
        if constexpr (sizeof(OutT) == 4) *p = vv; else *(unsigned short*)p = bfbits(vv);
      } else {
        OutT* p = out + (size_t)row * ldOut + col;
        if constexpr (sizeof(OutT) == 4) *p = vv; else *(unsigned short*)p = bfbits(vv);
      }
    }
  }
}

// ---------------------------------------------------------------------------
// LayerNorm(256) + exact-erf GELU; one wave per token row. f32 in, bf16 out.
// ---------------------------------------------------------------------------
__global__ void ln_gelu_kernel(const float* __restrict__ xs,
                               const float* __restrict__ gamma,
                               const float* __restrict__ beta,
                               unsigned short* __restrict__ xbf) {
  const int wave = (int)((blockIdx.x * blockDim.x + threadIdx.x) >> 5);
  if (wave >= B_ * M_) return;
  const int lane = threadIdx.x & 31;
  const float* row = xs + (size_t)wave * C_;
  float v[8], s = 0.0f;
#pragma unroll
  for (int j = 0; j < 8; ++j) { v[j] = row[lane + 32 * j]; s += v[j]; }
  const float mu = wave_sum(s) * (1.0f / C_);
  float vs = 0.0f;
#pragma unroll
  for (int j = 0; j < 8; ++j) { float d = v[j] - mu; vs += d * d; }
  const float var = wave_sum(vs) * (1.0f / C_);
  const float inv = rsqrtf(var + LN_EPS_);
#pragma unroll
  for (int j = 0; j < 8; ++j) {
    int c = lane + 32 * j;
    float t = (v[j] - mu) * inv * gamma[c] + beta[c];
    float g = 0.5f * t * (1.0f + erff(t * 0.70710678118654752f));
    xbf[(size_t)wave * C_ + c] = bfbits(g);
  }
}

// ---------------------------------------------------------------------------
// k[b,h,m] = sum_c x_[b,m,c] * Wk[h,c];  one wave per (b,m)
// ---------------------------------------------------------------------------
__global__ void k_kernel(const unsigned short* __restrict__ xbf,
                         const float* __restrict__ Wk,
                         float* __restrict__ kArr) {
  const int wave = (int)((blockIdx.x * blockDim.x + threadIdx.x) >> 5);
  if (wave >= B_ * M_) return;
  const int lane = threadIdx.x & 31;
  const int b = wave >> 10, m = wave & (M_ - 1);
  union { uint4 u; unsigned short s[8]; } xr;
  xr.u = *(const uint4*)(xbf + (size_t)wave * C_ + lane * 8);
  float xv[8];
#pragma unroll
  for (int i = 0; i < 8; ++i) xv[i] = bf2f(xr.s[i]);
#pragma unroll
  for (int h = 0; h < NHEADS; ++h) {
    const float* wk = Wk + h * C_ + lane * 8;
    float p = 0.0f;
#pragma unroll
    for (int i = 0; i < 8; ++i) p += xv[i] * wk[i];
    p = wave_sum(p);
    if (lane == 0) kArr[((size_t)b * NHEADS + h) * M_ + m] = p;
  }
}

// per-(b,h) k max/min for the rank-1 softmax row-max trick; one wave each
__global__ void kminmax_kernel(const float* __restrict__ kArr,
                               float* __restrict__ kmax,
                               float* __restrict__ kmin) {
  const int wave = (int)((blockIdx.x * blockDim.x + threadIdx.x) >> 5);
  if (wave >= B_ * NHEADS) return;
  const int lane = threadIdx.x & 31;
  const float* kp = kArr + (size_t)wave * M_;
  float mx = -3.0e38f, mn = 3.0e38f;
  for (int i = lane; i < M_; i += 32) {
    float t = kp[i];
    mx = fmaxf(mx, t);
    mn = fminf(mn, t);
  }
  mx = wave_max(mx);
  mn = wave_min(mn);
  if (lane == 0) { kmax[wave] = mx; kmin[wave] = mn; }
}

// ---------------------------------------------------------------------------
// q[b,h,n] = sum_c x[b,c,n] * Wq[h,c]; one thread per (b,n), all 8 heads
// ---------------------------------------------------------------------------
__global__ void q_kernel(const float* __restrict__ x,
                         const float* __restrict__ Wq,
                         float* __restrict__ qArr) {
  const int tid = blockIdx.x * blockDim.x + threadIdx.x;
  if (tid >= B_ * N_) return;
  const int n = tid & (N_ - 1);
  const int b = tid >> 12;
  const float* xb = x + (size_t)b * C_ * N_ + n;
  float acc[NHEADS] = {};
  for (int c = 0; c < C_; ++c) {
    float xv = xb[(size_t)c * N_];
#pragma unroll
    for (int h = 0; h < NHEADS; ++h) acc[h] += xv * Wq[h * C_ + c];
  }
#pragma unroll
  for (int h = 0; h < NHEADS; ++h)
    qArr[((size_t)b * NHEADS + h) * N_ + n] = acc[h];
}

// ---------------------------------------------------------------------------
// Rank-1 attention, flash-style:  out[b,h,n,:] = softmax_m(a_n k_m) · V
// One wave per (b, h, 32-row n-tile): TWO A fragments (rows n0..n0+15 and
// n0+16..n0+31) share the per-m k loads and the V B-fragments, so each
// 32-wide m-chunk costs 16 k loads + 4 b128 V loads + 32 exps for 6 WMMAs
// (2 d-subtiles x 2 row tiles + 2 ones-WMMAs giving both row-sum sets in
// D layout). Normalization is a lane-local v_rcp_f32 multiply.
// ---------------------------------------------------------------------------
__global__ void attn_kernel(const float* __restrict__ qArr,
                            const float* __restrict__ kArr,
                            const float* __restrict__ kmax,
                            const float* __restrict__ kmin,
                            const unsigned short* __restrict__ vT,   // [B, C, M]
                            unsigned short* __restrict__ att) {      // [B*N, C]
  const int wave = (int)((blockIdx.x * blockDim.x + threadIdx.x) >> 5);
  if (wave >= B_ * NHEADS * (N_ / 32)) return;
  const int ntile = wave & 127;          // N_/32 = 128
  const int h = (wave >> 7) & 7;
  const int b = wave >> 10;
  const int lane = threadIdx.x & 31;
  const int l16 = lane & 15;
  const bool hi = lane >= 16;

  // A-operand row for this lane is n = l16 within each 16-row tile
  const int bh = b * NHEADS + h;
  const int n0 = ntile * 32 + l16;
  const float a0 = qArr[(size_t)bh * N_ + n0] * SCALE_;
  const float a1 = qArr[(size_t)bh * N_ + n0 + 16] * SCALE_;
  const float km = kmax[bh], kn = kmin[bh];
  const float rmax0 = (a0 >= 0.0f) ? a0 * km : a0 * kn;
  const float rmax1 = (a1 >= 0.0f) ? a1 * km : a1 * kn;
  const float* kp = kArr + (size_t)bh * M_;

  const unsigned short* vp = vT + ((size_t)b * C_ + h * DHEAD) * M_;
  const int mb  = hi ? 8 : 0;    // A-frag K-offset base for this lane half
  const int kbB = hi ? 8 : 0;    // B-frag K-offset base

  Frag16 ones;
  ones.u4[0] = make_uint4(0x3F803F80u, 0x3F803F80u, 0x3F803F80u, 0x3F803F80u);
  ones.u4[1] = ones.u4[0];

  v8f acc00 = {}, acc01 = {}, accs0 = {};
  v8f acc10 = {}, acc11 = {}, accs1 = {};
  for (int mc = 0; mc < M_; mc += 32) {
    Frag16 p0, p1;
#pragma unroll
    for (int i = 0; i < 16; ++i) {
      // A 16x32 bf16 layout: element i of lane -> K = mb + (i&7) + ((i>>3)<<4)
      const int moff = mb + (i & 7) + ((i >> 3) << 4);
      const float kv = kp[mc + moff];                 // shared by both row tiles
      p0.v[i] = (__bf16)__expf(a0 * kv - rmax0);
      p1.v[i] = (__bf16)__expf(a1 * kv - rmax1);
    }
    // V chunk as B-operand: lane = column d, m values kbB+0..7 and kbB+16..23 (vT layout)
    const unsigned short* c0 = vp + (size_t)l16 * M_ + mc + kbB;
    const unsigned short* c1 = vp + (size_t)(16 + l16) * M_ + mc + kbB;
    Frag16 v0, v1;
    v0.u4[0] = *(const uint4*)c0;  v0.u4[1] = *(const uint4*)(c0 + 16);
    v1.u4[0] = *(const uint4*)c1;  v1.u4[1] = *(const uint4*)(c1 + 16);
    acc00 = __builtin_amdgcn_wmma_f32_16x16x32_bf16(false, p0.v, false, v0.v,
                                                    (short)0, acc00, false, false);
    acc01 = __builtin_amdgcn_wmma_f32_16x16x32_bf16(false, p0.v, false, v1.v,
                                                    (short)0, acc01, false, false);
    accs0 = __builtin_amdgcn_wmma_f32_16x16x32_bf16(false, p0.v, false, ones.v,
                                                    (short)0, accs0, false, false);
    acc10 = __builtin_amdgcn_wmma_f32_16x16x32_bf16(false, p1.v, false, v0.v,
                                                    (short)0, acc10, false, false);
    acc11 = __builtin_amdgcn_wmma_f32_16x16x32_bf16(false, p1.v, false, v1.v,
                                                    (short)0, acc11, false, false);
    accs1 = __builtin_amdgcn_wmma_f32_16x16x32_bf16(false, p1.v, false, ones.v,
                                                    (short)0, accs1, false, false);
  }
#pragma unroll
  for (int r = 0; r < 8; ++r) {
    const int rr = r + (hi ? 8 : 0);                  // D layout row
    // denominators are positive & well-scaled: fast v_rcp_f32 is sufficient
    const float inv0 = __builtin_amdgcn_rcpf(accs0[r]);
    const float inv1 = __builtin_amdgcn_rcpf(accs1[r]);
    const size_t base0 = ((size_t)b * N_ + ntile * 32 + rr) * C_ + h * DHEAD;
    const size_t base1 = base0 + (size_t)16 * C_;
    att[base0 + l16]      = bfbits(acc00[r] * inv0);
    att[base0 + 16 + l16] = bfbits(acc01[r] * inv0);
    att[base1 + l16]      = bfbits(acc10[r] * inv1);
    att[base1 + 16 + l16] = bfbits(acc11[r] * inv1);
  }
}

// ---------------------------------------------------------------------------
extern "C" void kernel_launch(void* const* d_in, const int* in_sizes, int n_in,
                              void* d_out, int out_size, void* d_ws, size_t ws_size,
                              hipStream_t stream) {
  (void)in_sizes; (void)n_in; (void)out_size; (void)ws_size;
  const float* x     = (const float*)d_in[0];
  const float* Wq    = (const float*)d_in[1];
  const float* Wk    = (const float*)d_in[2];
  const float* Wv    = (const float*)d_in[3];
  const float* Wsr   = (const float*)d_in[4];
  const float* bsr   = (const float*)d_in[5];
  const float* gamma = (const float*)d_in[6];
  const float* beta  = (const float*)d_in[7];
  const float* Wp    = (const float*)d_in[8];
  const float* bp    = (const float*)d_in[9];
  float* out = (float*)d_out;

  char* w = (char*)d_ws;
  auto alloc = [&](size_t bytes) -> char* {
    char* p = w; w += (bytes + 255) & ~(size_t)255; return p;
  };
  unsigned short* WsrBf  = (unsigned short*)alloc((size_t)C_ * C_ * 2);
  unsigned short* WvBf   = (unsigned short*)alloc((size_t)C_ * C_ * 2);
  unsigned short* WpBf   = (unsigned short*)alloc((size_t)C_ * C_ * 2);
  unsigned short* poolBf = (unsigned short*)alloc((size_t)B_ * M_ * C_ * 2);
  float*          xsF    = (float*)         alloc((size_t)B_ * M_ * C_ * 4);
  unsigned short* xBf    = (unsigned short*)alloc((size_t)B_ * M_ * C_ * 2);
  float*          kArr   = (float*)         alloc((size_t)B_ * NHEADS * M_ * 4);
  float*          qArr   = (float*)         alloc((size_t)B_ * NHEADS * N_ * 4);
  float*          kmaxA  = (float*)         alloc(B_ * NHEADS * 4);
  float*          kminA  = (float*)         alloc(B_ * NHEADS * 4);
  unsigned short* vT     = (unsigned short*)alloc((size_t)B_ * C_ * M_ * 2);
  unsigned short* attBf  = (unsigned short*)alloc((size_t)B_ * N_ * C_ * 2);

  const dim3 blk(256);

  // weights -> bf16
  f32_to_bf16_kernel<<<(C_ * C_) / 256, blk, 0, stream>>>(Wsr, WsrBf, C_ * C_);
  f32_to_bf16_kernel<<<(C_ * C_) / 256, blk, 0, stream>>>(Wv,  WvBf,  C_ * C_);
  f32_to_bf16_kernel<<<(C_ * C_) / 256, blk, 0, stream>>>(Wp,  WpBf,  C_ * C_);

  // q (independent of pooled path)
  q_kernel<<<(B_ * N_) / 256, blk, 0, stream>>>(x, Wq, qArr);

  // pool -> tokens
  pool_kernel<<<(B_ * C_ * M_) / 256, blk, 0, stream>>>(x, poolBf);

  // spatial-reduction 1x1 conv GEMM: xs = pool · Wsrᵀ + bsr   (f32 out)
  {
    int rows = B_ * M_, cols = C_;
    int waves = (rows / 16) * (cols / 64);
    gemm_bf16_wmma<float, false, true><<<waves * 32 / 256, blk, 0, stream>>>(
        poolBf, WsrBf, bsr, xsF, rows, C_, cols, rows, 0LL, C_);
  }

  // LayerNorm + GELU
  ln_gelu_kernel<<<(B_ * M_) / 8, blk, 0, stream>>>(xsF, gamma, beta, xBf);

  // k, kmax/kmin
  k_kernel<<<(B_ * M_) / 8, blk, 0, stream>>>(xBf, Wk, kArr);
  kminmax_kernel<<<(B_ * NHEADS + 7) / 8, blk, 0, stream>>>(kArr, kmaxA, kminA);

  // v = x_ · Wvᵀ, stored TRANSPOSED as vT[b][o][m] so attention B-frags are contiguous
  {
    int rows = B_ * M_, cols = C_;
    int waves = (rows / 16) * (cols / 64);
    gemm_bf16_wmma<unsigned short, true, false><<<waves * 32 / 256, blk, 0, stream>>>(
        xBf, WvBf, nullptr, vT, rows, C_, cols, M_, (long long)C_ * M_, M_);
  }

  // rank-1 softmax attention + P·V via WMMA (32 query rows per wave)
  attn_kernel<<<(B_ * NHEADS * (N_ / 32)) / 8, blk, 0, stream>>>(
      qArr, kArr, kmaxA, kminA, vT, attBf);

  // output projection: out = att · Wpᵀ + bp, stored transposed -> [B, C, N]
  {
    int rows = B_ * N_, cols = C_;
    int waves = (rows / 16) * (cols / 64);
    gemm_bf16_wmma<float, true, true><<<waves * 32 / 256, blk, 0, stream>>>(
        attBf, WpBf, bp, out, rows, C_, cols, N_, (long long)C_ * N_, N_);
  }
}
